// AttentionActor_23278722744394
// MI455X (gfx1250) — matrix operand order repaired and checked by reading, not compiled, not run on hardware
//
#include <hip/hip_runtime.h>
#include <stdint.h>
#include <stddef.h>

// Problem constants (from reference setup_inputs)
#define NB 256
#define NT 4
#define ND 16
#define NE 64
#define NH 4
#define NHD 16
#define NF 4096
#define WAVES 8
#define TPW 32          // tiles per wave: 4096 / (8 waves * 16 fires)

typedef __attribute__((ext_vector_type(16))) _Float16 v16h;
typedef __attribute__((ext_vector_type(8)))  float    v8f;

union H16 { v16h v; _Float16 e[16]; };

struct Args {
  const float* dpos; const float* gpos; const float* fpos;
  const unsigned char* masks;   // jnp bool -> 1 byte each (all False in setup)
  const int* acts;
  const float *pos_w,*pos_b,*goal_w,*goal_b,*fire_w,*fire_b,*vel_w,*vel_b;
  const float *id_emb,*f1_w,*f1_b,*f2_w,*f2_b;
  const float *sa_wq,*sa_bq,*sa_wk,*sa_bk,*sa_wv,*sa_bv,*sa_wo,*sa_bo,*sa_g,*sa_be;
  const float *ca_wq,*ca_bq,*ca_wk,*ca_bk,*ca_wv,*ca_bv,*ca_wo,*ca_bo,*ca_g,*ca_be;
  const float *qp_w,*qp_b,*kp_w,*kp_b,*logT,*logP;
  float* out;   // [0,4096): actions as float, [4096,8192): log_probs, [8192,12288): entropy
};

// A-fragment K index for 16-bit 16x32 A matrix (lane-group g): contiguous k = 16*g + e (matches B)
__device__ __forceinline__ int kmapA(int e, int g) { return 16*g + e; }

__global__ __launch_bounds__(256) void actor_kernel(Args a) {
  __shared__ float s_x[ND][4*NE];        // concat embeddings
  __shared__ float s_h[ND][2*NE];
  __shared__ float s_emb[ND][NE];
  __shared__ float s_emb2[ND][NE];       // post self-attn LN
  __shared__ float s_repr[ND][NE];       // post cross-attn LN
  __shared__ float s_q[ND][NE];
  __shared__ float s_k[ND][NE];
  __shared__ float s_v[ND][NE];
  __shared__ float s_qca[ND][NE];        // ca queries, pre-scaled 1/sqrt(HD)
  __shared__ float s_qfin[ND][NE];       // final q, pre-scaled 1/(8*temp)
  __shared__ float s_ctx[ND][NE];        // ca context
  __shared__ float s_fw[2*NE];
  __shared__ float s_fb[NE];
  __shared__ float s_tile[WAVES][16][NE];   // per-wave k / v / kq tile (fires x feat)
  __shared__ float s_lt[WAVES][16][16];     // per-wave logits tile (drone x fire)
  __shared__ unsigned char s_mt[WAVES][16];
  __shared__ float s_pm[WAVES][64];
  __shared__ float s_ps[WAVES][64];
  __shared__ float s_pa[WAVES][64][16];
  __shared__ float s_p2[WAVES][2][16][3];
  __shared__ float s_fe[ND][NE];
  __shared__ float s_kqA[ND][NE];
  __shared__ float s_alog[ND][ND];
  __shared__ int   s_act[ND];

  const int tid  = threadIdx.x;
  const int b    = blockIdx.x;
  const int lane = tid & 31, wid = tid >> 5;
  const int nl   = lane & 15, gg = lane >> 4;

  float temp = fmaxf(__expf(a.logT[0]), 0.01f);
  const float qscale = 1.0f / (8.0f * temp);          // 1/(sqrt(E)*temp)
  const float pen = fmaxf(__expf(a.logP[0]), 0.1f);

  const float* fprow = a.fpos + ((size_t)(b*NT + 3)) * NF * 2;
  const unsigned char* mrow = a.masks + (size_t)b * NF;

  for (int i = tid; i < 2*NE; i += 256) s_fw[i] = a.fire_w[i];
  for (int i = tid; i < NE;   i += 256) s_fb[i] = a.fire_b[i];
  if (tid < ND) {
    int ac = a.acts[b*ND + tid];
    s_act[tid] = ac;
    a.out[b*ND + tid] = (float)ac;    // output 0: actions_idx
  }

  // ---------------- Stage 1: drone embeddings (scalar, tiny) ----------------
  for (int i = tid; i < ND*NE; i += 256) {
    int n = i >> 6, c = i & 63;
    const float* dp3 = a.dpos + ((b*NT + 3)*ND + n)*2;
    const float* dp2 = a.dpos + ((b*NT + 2)*ND + n)*2;
    const float* gp3 = a.gpos + ((b*NT + 3)*ND + n)*2;
    float px = dp3[0], py = dp3[1];
    float vx = px - dp2[0], vy = py - dp2[1];
    float gx = gp3[0], gy = gp3[1];
    s_x[n][c]       = fmaxf(px*a.pos_w[c]  + py*a.pos_w[64+c]  + a.pos_b[c],  0.f);
    s_x[n][64+c]    = fmaxf(gx*a.goal_w[c] + gy*a.goal_w[64+c] + a.goal_b[c], 0.f);
    s_x[n][128+c]   = a.id_emb[n*64 + c];
    s_x[n][192+c]   = fmaxf(vx*a.vel_w[c]  + vy*a.vel_w[64+c]  + a.vel_b[c],  0.f);
  }
  __syncthreads();

  { // h = relu(x @ f1_w + f1_b)  (16x256)@(256x128)
    int j = tid & 127, nh = tid >> 7;
    float acc[8];
#pragma unroll
    for (int r = 0; r < 8; ++r) acc[r] = a.f1_b[j];
    for (int k = 0; k < 256; ++k) {
      float w = a.f1_w[k*128 + j];
#pragma unroll
      for (int r = 0; r < 8; ++r) acc[r] += s_x[nh*8 + r][k] * w;
    }
#pragma unroll
    for (int r = 0; r < 8; ++r) s_h[nh*8 + r][j] = fmaxf(acc[r], 0.f);
  }
  __syncthreads();

  { // emb = relu(h @ f2_w + f2_b)
    int j = tid & 63, grp = tid >> 6;
    float acc[4];
#pragma unroll
    for (int r = 0; r < 4; ++r) acc[r] = a.f2_b[j];
    for (int k = 0; k < 128; ++k) {
      float w = a.f2_w[k*64 + j];
#pragma unroll
      for (int r = 0; r < 4; ++r) acc[r] += s_h[grp*4 + r][k] * w;
    }
#pragma unroll
    for (int r = 0; r < 4; ++r) s_emb[grp*4 + r][j] = fmaxf(acc[r], 0.f);
  }
  __syncthreads();

  { // self-attention q,k,v projections
    int j = tid & 63, grp = tid >> 6;
    float aq[4], ak[4], av[4];
#pragma unroll
    for (int r = 0; r < 4; ++r) { aq[r] = a.sa_bq[j]; ak[r] = a.sa_bk[j]; av[r] = a.sa_bv[j]; }
    for (int k = 0; k < 64; ++k) {
      float wq = a.sa_wq[k*64 + j], wk = a.sa_wk[k*64 + j], wv2 = a.sa_wv[k*64 + j];
#pragma unroll
      for (int r = 0; r < 4; ++r) {
        float e = s_emb[grp*4 + r][k];
        aq[r] += e*wq; ak[r] += e*wk; av[r] += e*wv2;
      }
    }
#pragma unroll
    for (int r = 0; r < 4; ++r) {
      s_q[grp*4 + r][j] = aq[r]; s_k[grp*4 + r][j] = ak[r]; s_v[grp*4 + r][j] = av[r];
    }
  }
  __syncthreads();

  { // self-attention (N=16, per (head,drone) thread)
    float o_reg[16];
    int hh = tid >> 4, dd = tid & 15;
    if (tid < 64) {
      float sc[16]; float mx = -INFINITY;
#pragma unroll
      for (int j = 0; j < 16; ++j) {
        float t = 0.f;
#pragma unroll
        for (int k = 0; k < 16; ++k) t += s_q[dd][hh*16 + k] * s_k[j][hh*16 + k];
        sc[j] = t * 0.25f; mx = fmaxf(mx, sc[j]);
      }
      float ss = 0.f;
#pragma unroll
      for (int j = 0; j < 16; ++j) { sc[j] = __expf(sc[j] - mx); ss += sc[j]; }
      float inv = 1.f / ss;
#pragma unroll
      for (int k = 0; k < 16; ++k) {
        float t = 0.f;
#pragma unroll
        for (int j = 0; j < 16; ++j) t += sc[j] * s_v[j][hh*16 + k];
        o_reg[k] = t * inv;
      }
    }
    __syncthreads();
    if (tid < 64)
#pragma unroll
      for (int k = 0; k < 16; ++k) s_q[dd][hh*16 + k] = o_reg[k];   // reuse s_q as sa output
  }
  __syncthreads();

  { // sa out-proj + residual
    int j = tid & 63, grp = tid >> 6;
    float acc[4];
#pragma unroll
    for (int r = 0; r < 4; ++r) acc[r] = a.sa_bo[j];
    for (int k = 0; k < 64; ++k) {
      float w = a.sa_wo[k*64 + j];
#pragma unroll
      for (int r = 0; r < 4; ++r) acc[r] += s_q[grp*4 + r][k] * w;
    }
#pragma unroll
    for (int r = 0; r < 4; ++r) s_emb2[grp*4 + r][j] = s_emb[grp*4 + r][j] + acc[r];
  }
  __syncthreads();
  if (tid < 16) { // LayerNorm
    float mu = 0.f;
    for (int c = 0; c < 64; ++c) mu += s_emb2[tid][c];
    mu *= (1.f/64.f);
    float var = 0.f;
    for (int c = 0; c < 64; ++c) { float t = s_emb2[tid][c] - mu; var += t*t; }
    float inv = rsqrtf(var*(1.f/64.f) + 1e-5f);
    for (int c = 0; c < 64; ++c)
      s_emb2[tid][c] = (s_emb2[tid][c] - mu)*inv*a.sa_g[c] + a.sa_be[c];
  }
  __syncthreads();

  { // ca query projection (pre-scaled by 1/sqrt(HD)=0.25)
    int j = tid & 63, grp = tid >> 6;
    float acc[4];
#pragma unroll
    for (int r = 0; r < 4; ++r) acc[r] = a.ca_bq[j];
    for (int k = 0; k < 64; ++k) {
      float w = a.ca_wq[k*64 + j];
#pragma unroll
      for (int r = 0; r < 4; ++r) acc[r] += s_emb2[grp*4 + r][k] * w;
    }
#pragma unroll
    for (int r = 0; r < 4; ++r) s_qca[grp*4 + r][j] = acc[r] * 0.25f;
  }
  __syncthreads();

  // ------------- Pass 1: cross-attention over fires (flash style) -------------
  // NOTE: each wave works only on its own s_tile[wid]/s_mt[wid] slice. LDS ops
  // from one wave complete in order (CDNA5 ISA), so cross-lane communication
  // within the wave needs NO block barrier -> waves stream fires independently.
  H16 bkl[4], bkh[4], bvl[4], bvh[4];
  float bkb[4], bvb[4];
#pragma unroll
  for (int cb = 0; cb < 4; ++cb) {
    int col = cb*16 + nl;
#pragma unroll
    for (int e = 0; e < 16; ++e) {
      int kb = 16*gg + e;
      bkl[cb].e[e] = (_Float16)a.ca_wk[kb*64 + col];
      bkh[cb].e[e] = (_Float16)a.ca_wk[(32 + kb)*64 + col];
      bvl[cb].e[e] = (_Float16)a.ca_wv[kb*64 + col];
      bvh[cb].e[e] = (_Float16)a.ca_wv[(32 + kb)*64 + col];
    }
    bkb[cb] = a.ca_bk[col]; bvb[cb] = a.ca_bv[col];
  }

  float m_[2] = {-INFINITY, -INFINITY}, sum_[2] = {0.f, 0.f};
  float accv[2][16] = {};
  float pw[2][16];

  for (int t = 0; t < TPW; ++t) {
    int fb = wid*512 + t*16;
    if (lane < 16) s_mt[wid][lane] = mrow[fb + lane];
    __builtin_prefetch(fprow + (fb + 16)*2, 0, 1);

    // fire_e A fragments computed in-register (2->64 projection)
    float px = fprow[(fb + nl)*2 + 0], py = fprow[(fb + nl)*2 + 1];
    H16 al, ah;
#pragma unroll
    for (int e = 0; e < 16; ++e) {
      int c = kmapA(e, gg);
      al.e[e] = (_Float16)fmaxf(px*s_fw[c]      + py*s_fw[64 + c]      + s_fb[c],      0.f);
      ah.e[e] = (_Float16)fmaxf(px*s_fw[32 + c] + py*s_fw[96 + c]      + s_fb[32 + c], 0.f);
    }

    // K tile = fire_e @ ca_wk + bk   (WMMA, K=64 via two chained 16x16x32)
#pragma unroll
    for (int cb = 0; cb < 4; ++cb) {
      v8f d = {};
      d = __builtin_amdgcn_wmma_f32_16x16x32_f16(false, al.v, false, bkl[cb].v, (short)0, d, false, false);
      d = __builtin_amdgcn_wmma_f32_16x16x32_f16(false, ah.v, false, bkh[cb].v, (short)0, d, false, false);
#pragma unroll
      for (int i = 0; i < 8; ++i) s_tile[wid][i + 8*gg][cb*16 + nl] = d[i] + bkb[cb];
    }

    // online-softmax score update; lane owns 2 (head,drone) rows
#pragma unroll
    for (int rr = 0; rr < 2; ++rr) {
      int row = lane + 32*rr; int hh = row >> 4, dd = row & 15;
      float sc[16]; float mx = m_[rr];
#pragma unroll
      for (int j = 0; j < 16; ++j) {
        if (!s_mt[wid][j]) {
          float t2 = 0.f;
#pragma unroll
          for (int k2 = 0; k2 < 16; ++k2) t2 += s_qca[dd][hh*16 + k2] * s_tile[wid][j][hh*16 + k2];
          sc[j] = t2; mx = fmaxf(mx, t2);
        } else sc[j] = -INFINITY;
      }
      if (mx > -INFINITY) {
        float scale = (m_[rr] > -INFINITY) ? __expf(m_[rr] - mx) : 0.f;
        m_[rr] = mx;
        float ss = 0.f;
#pragma unroll
        for (int j = 0; j < 16; ++j) {
          float pj = (sc[j] > -INFINITY) ? __expf(sc[j] - mx) : 0.f;
          pw[rr][j] = pj; ss += pj;
        }
        sum_[rr] = sum_[rr]*scale + ss;
#pragma unroll
        for (int k2 = 0; k2 < 16; ++k2) accv[rr][k2] *= scale;
      } else {
#pragma unroll
        for (int j = 0; j < 16; ++j) pw[rr][j] = 0.f;
      }
    }

    // V tile = fire_e @ ca_wv + bv (in-wave LDS ordering keeps WAR safe)
#pragma unroll
    for (int cb = 0; cb < 4; ++cb) {
      v8f d = {};
      d = __builtin_amdgcn_wmma_f32_16x16x32_f16(false, al.v, false, bvl[cb].v, (short)0, d, false, false);
      d = __builtin_amdgcn_wmma_f32_16x16x32_f16(false, ah.v, false, bvh[cb].v, (short)0, d, false, false);
#pragma unroll
      for (int i = 0; i < 8; ++i) s_tile[wid][i + 8*gg][cb*16 + nl] = d[i] + bvb[cb];
    }

#pragma unroll
    for (int rr = 0; rr < 2; ++rr) {
      int row = lane + 32*rr; int hh = row >> 4;
#pragma unroll
      for (int k2 = 0; k2 < 16; ++k2) {
        float t2 = 0.f;
#pragma unroll
        for (int j = 0; j < 16; ++j) t2 += pw[rr][j] * s_tile[wid][j][hh*16 + k2];
        accv[rr][k2] += t2;
      }
    }
  }

  // merge 8 per-wave flash partials (cross-wave -> real barrier)
#pragma unroll
  for (int rr = 0; rr < 2; ++rr) {
    int row = lane + 32*rr;
    s_pm[wid][row] = m_[rr]; s_ps[wid][row] = sum_[rr];
#pragma unroll
    for (int k2 = 0; k2 < 16; ++k2) s_pa[wid][row][k2] = accv[rr][k2];
  }
  __syncthreads();
  if (tid < 64) {
    int row = tid, hh = row >> 4, dd = row & 15;
    float M = -INFINITY;
    for (int w = 0; w < WAVES; ++w) M = fmaxf(M, s_pm[w][row]);
    if (M > -INFINITY) {
      float wgt[WAVES]; float S = 0.f;
      for (int w = 0; w < WAVES; ++w) {
        wgt[w] = (s_pm[w][row] > -INFINITY) ? __expf(s_pm[w][row] - M) : 0.f;
        S += s_ps[w][row] * wgt[w];
      }
      float invS = 1.f / S;
      for (int k2 = 0; k2 < 16; ++k2) {
        float t2 = 0.f;
        for (int w = 0; w < WAVES; ++w) t2 += wgt[w] * s_pa[w][row][k2];
        s_ctx[dd][hh*16 + k2] = t2 * invS;
      }
    } else {
      for (int k2 = 0; k2 < 16; ++k2) s_ctx[dd][hh*16 + k2] = 0.f;
    }
  }
  __syncthreads();

  { // ca out-proj + residual
    int j = tid & 63, grp = tid >> 6;
    float acc[4];
#pragma unroll
    for (int r = 0; r < 4; ++r) acc[r] = a.ca_bo[j];
    for (int k = 0; k < 64; ++k) {
      float w = a.ca_wo[k*64 + j];
#pragma unroll
      for (int r = 0; r < 4; ++r) acc[r] += s_ctx[grp*4 + r][k] * w;
    }
#pragma unroll
    for (int r = 0; r < 4; ++r) s_repr[grp*4 + r][j] = s_emb2[grp*4 + r][j] + acc[r];
  }
  __syncthreads();
  if (tid < 16) { // LayerNorm
    float mu = 0.f;
    for (int c = 0; c < 64; ++c) mu += s_repr[tid][c];
    mu *= (1.f/64.f);
    float var = 0.f;
    for (int c = 0; c < 64; ++c) { float t = s_repr[tid][c] - mu; var += t*t; }
    float inv = rsqrtf(var*(1.f/64.f) + 1e-5f);
    for (int c = 0; c < 64; ++c)
      s_repr[tid][c] = (s_repr[tid][c] - mu)*inv*a.ca_g[c] + a.ca_be[c];
  }
  __syncthreads();

  { // final q projection, pre-scaled by 1/(8*temp)
    int j = tid & 63, grp = tid >> 6;
    float acc[4];
#pragma unroll
    for (int r = 0; r < 4; ++r) acc[r] = a.qp_b[j];
    for (int k = 0; k < 64; ++k) {
      float w = a.qp_w[k*64 + j];
#pragma unroll
      for (int r = 0; r < 4; ++r) acc[r] += s_repr[grp*4 + r][k] * w;
    }
#pragma unroll
    for (int r = 0; r < 4; ++r) s_qfin[grp*4 + r][j] = acc[r] * qscale;
  }
  __syncthreads();

  // exact logits at the <=16 action fires (for penalty corrections & log_probs)
  for (int i = tid; i < ND*NE; i += 256) {
    int jj = i >> 6, c = i & 63;
    int fi = s_act[jj];
    float px = fprow[fi*2], py = fprow[fi*2 + 1];
    s_fe[jj][c] = fmaxf(px*s_fw[c] + py*s_fw[64 + c] + s_fb[c], 0.f);
  }
  __syncthreads();
  for (int i = tid; i < ND*NE; i += 256) {
    int jj = i >> 6, c = i & 63;
    float acc = a.kp_b[c];
    for (int e = 0; e < 64; ++e) acc += s_fe[jj][e] * a.kp_w[e*64 + c];
    s_kqA[jj][c] = acc;
  }
  __syncthreads();
  {
    int dd = tid >> 4, jj = tid & 15;
    float acc = 0.f;
    for (int k = 0; k < 64; ++k) acc += s_qfin[dd][k] * s_kqA[jj][k];
    s_alog[dd][jj] = acc;
  }
  __syncthreads();

  // ------------- Pass 2: logits over all fires + streaming log-softmax -------------
  H16 qal, qah;
#pragma unroll
  for (int e = 0; e < 16; ++e) {
    int c = kmapA(e, gg);
    qal.e[e] = (_Float16)s_qfin[nl][c];
    qah.e[e] = (_Float16)s_qfin[nl][32 + c];
  }
  H16 bpl[4], bph[4]; float bpb[4];
#pragma unroll
  for (int cb = 0; cb < 4; ++cb) {
    int col = cb*16 + nl;
#pragma unroll
    for (int e = 0; e < 16; ++e) {
      int kb = 16*gg + e;
      bpl[cb].e[e] = (_Float16)a.kp_w[kb*64 + col];
      bph[cb].e[e] = (_Float16)a.kp_w[(32 + kb)*64 + col];
    }
    bpb[cb] = a.kp_b[col];
  }

  float M2 = -INFINITY, S2 = 0.f, T2 = 0.f;
  for (int t = 0; t < TPW; ++t) {
    int fb = wid*512 + t*16;
    if (lane < 16) s_mt[wid][lane] = mrow[fb + lane];
    __builtin_prefetch(fprow + (fb + 16)*2, 0, 1);

    float px = fprow[(fb + nl)*2 + 0], py = fprow[(fb + nl)*2 + 1];
    H16 al, ah;
#pragma unroll
    for (int e = 0; e < 16; ++e) {
      int c = kmapA(e, gg);
      al.e[e] = (_Float16)fmaxf(px*s_fw[c]      + py*s_fw[64 + c] + s_fb[c],      0.f);
      ah.e[e] = (_Float16)fmaxf(px*s_fw[32 + c] + py*s_fw[96 + c] + s_fb[32 + c], 0.f);
    }
    // kq tile = fire_e @ kp_w + kp_b
#pragma unroll
    for (int cb = 0; cb < 4; ++cb) {
      v8f d = {};
      d = __builtin_amdgcn_wmma_f32_16x16x32_f16(false, al.v, false, bpl[cb].v, (short)0, d, false, false);
      d = __builtin_amdgcn_wmma_f32_16x16x32_f16(false, ah.v, false, bph[cb].v, (short)0, d, false, false);
#pragma unroll
      for (int i = 0; i < 8; ++i) s_tile[wid][i + 8*gg][cb*16 + nl] = d[i] + bpb[cb];
    }

    // logits tile = q(16x64) @ kq^T(64x16): transpose kq via LDS into B fragments
    // (in-wave LDS ordering guarantees the stores above are visible)
    H16 t0, t1;
#pragma unroll
    for (int e = 0; e < 16; ++e) {
      int kb = 16*gg + e;
      t0.e[e] = (_Float16)s_tile[wid][nl][kb];
      t1.e[e] = (_Float16)s_tile[wid][nl][32 + kb];
    }
    v8f lg = {};
    lg = __builtin_amdgcn_wmma_f32_16x16x32_f16(false, qal.v, false, t0.v, (short)0, lg, false, false);
    lg = __builtin_amdgcn_wmma_f32_16x16x32_f16(false, qah.v, false, t1.v, (short)0, lg, false, false);
#pragma unroll
    for (int i = 0; i < 8; ++i) s_lt[wid][i + 8*gg][nl] = lg[i];

    // streaming (M, sum e^l, sum l*e^l); lane (nl,gg) scans drone nl, fires gg::2
    {
      int dd = nl;
#pragma unroll
      for (int jj2 = 0; jj2 < 8; ++jj2) {
        int jj = gg + 2*jj2;
        if (!s_mt[wid][jj]) {
          float l = s_lt[wid][dd][jj];
          if (l > M2) {
            float c2 = (M2 > -INFINITY) ? __expf(M2 - l) : 0.f;
            S2 *= c2; T2 *= c2; M2 = l;
          }
          float p = __expf(l - M2);
          S2 += p; T2 += l*p;
        }
      }
    }
  }
  s_p2[wid][gg][nl][0] = M2;
  s_p2[wid][gg][nl][1] = S2;
  s_p2[wid][gg][nl][2] = T2;
  __syncthreads();

  // ------------- Final: merge, penalty corrections, outputs -------------
  if (tid < 16) {
    int dd = tid;
    float M = -INFINITY;
    for (int w = 0; w < WAVES; ++w)
      for (int g2 = 0; g2 < 2; ++g2) M = fmaxf(M, s_p2[w][g2][dd][0]);
    float S = 0.f, T = 0.f;
    for (int w = 0; w < WAVES; ++w)
      for (int g2 = 0; g2 < 2; ++g2) {
        float mm = s_p2[w][g2][dd][0];
        if (mm > -INFINITY) {
          float c2 = __expf(mm - M);
          S += s_p2[w][g2][dd][1] * c2;
          T += s_p2[w][g2][dd][2] * c2;
        }
      }
    // sparse penalty corrections: logits[f] -> logits[f] - pen*count for fires
    // chosen by earlier drones (distinct fires only)
    for (int j = 0; j < 16; ++j) {
      int f = s_act[j];
      bool dup = false;
      for (int j2 = 0; j2 < j; ++j2) if (s_act[j2] == f) dup = true;
      if (dup) continue;
      int cnt = 0;
      for (int n2 = 0; n2 < dd; ++n2) if (s_act[n2] == f) cnt++;
      if (cnt == 0) continue;
      if (mrow[f]) continue;   // masked -> -inf either way
      float l  = s_alog[dd][j];
      float lp = l - pen * (float)cnt;
      float e0 = __expf(l - M), e1 = __expf(lp - M);
      S += e1 - e0;
      T += lp*e1 - l*e0;
    }
    float logZ = M + __logf(S);
    float ent  = logZ - T / S;
    int aa = s_act[dd];
    int cs = 0;
    for (int n2 = 0; n2 < dd; ++n2) if (s_act[n2] == aa) cs++;
    float la = s_alog[dd][dd] - pen * (float)cs;
    float lpout = mrow[aa] ? -INFINITY : (la - logZ);
    a.out[NB*ND + b*ND + dd]   = lpout;   // log_probs
    a.out[2*NB*ND + b*ND + dd] = ent;     // entropy
  }
}

extern "C" void kernel_launch(void* const* d_in, const int* in_sizes, int n_in,
                              void* d_out, int out_size, void* d_ws, size_t ws_size,
                              hipStream_t stream) {
  (void)in_sizes; (void)n_in; (void)d_ws; (void)ws_size; (void)out_size;
  Args a;
  a.dpos  = (const float*)d_in[0];
  a.gpos  = (const float*)d_in[1];
  a.fpos  = (const float*)d_in[2];
  a.masks = (const unsigned char*)d_in[3];
  a.acts  = (const int*)d_in[4];
  a.pos_w  = (const float*)d_in[5];  a.pos_b  = (const float*)d_in[6];
  a.goal_w = (const float*)d_in[7];  a.goal_b = (const float*)d_in[8];
  a.fire_w = (const float*)d_in[9];  a.fire_b = (const float*)d_in[10];
  a.vel_w  = (const float*)d_in[11]; a.vel_b  = (const float*)d_in[12];
  a.id_emb = (const float*)d_in[13];
  a.f1_w = (const float*)d_in[14]; a.f1_b = (const float*)d_in[15];
  a.f2_w = (const float*)d_in[16]; a.f2_b = (const float*)d_in[17];
  a.sa_wq = (const float*)d_in[18]; a.sa_bq = (const float*)d_in[19];
  a.sa_wk = (const float*)d_in[20]; a.sa_bk = (const float*)d_in[21];
  a.sa_wv = (const float*)d_in[22]; a.sa_bv = (const float*)d_in[23];
  a.sa_wo = (const float*)d_in[24]; a.sa_bo = (const float*)d_in[25];
  a.sa_g  = (const float*)d_in[26]; a.sa_be = (const float*)d_in[27];
  a.ca_wq = (const float*)d_in[28]; a.ca_bq = (const float*)d_in[29];
  a.ca_wk = (const float*)d_in[30]; a.ca_bk = (const float*)d_in[31];
  a.ca_wv = (const float*)d_in[32]; a.ca_bv = (const float*)d_in[33];
  a.ca_wo = (const float*)d_in[34]; a.ca_bo = (const float*)d_in[35];
  a.ca_g  = (const float*)d_in[36]; a.ca_be = (const float*)d_in[37];
  a.qp_w = (const float*)d_in[38]; a.qp_b = (const float*)d_in[39];
  a.kp_w = (const float*)d_in[40]; a.kp_b = (const float*)d_in[41];
  a.logT = (const float*)d_in[42]; a.logP = (const float*)d_in[43];
  a.out = (float*)d_out;
  actor_kernel<<<dim3(NB), dim3(256), 0, stream>>>(a);
}
